// MixedSparseGatedMLP_82248623718483
// MI455X (gfx1250) — matrix operand order repaired and checked by reading, not compiled, not run on hardware
//
#include <hip/hip_runtime.h>

// ---------------------------------------------------------------------------
// MixedSparseGatedMLP on gfx1250 (MI455X): bf16 WMMA + async-to-LDS pipeline
//   M = B*S = 8192 tokens, D = 2048, F = 8192, R = 16
// ---------------------------------------------------------------------------

#define M_TOK 8192
#define DIM   2048
#define FF    8192
#define RLORA 16

typedef __bf16 bf16;
typedef __attribute__((ext_vector_type(8)))  __bf16 v8bf;
typedef __attribute__((ext_vector_type(16))) __bf16 v16bf;
typedef __attribute__((ext_vector_type(8)))  float  v8f;

#if __has_builtin(__builtin_amdgcn_global_load_async_to_lds_b128)
#define HAVE_ASYNC 1
#endif

static __device__ inline v16bf combine16(v8bf lo, v8bf hi) {
  v16bf r;
#pragma unroll
  for (int i = 0; i < 8; ++i) { r[i] = lo[i]; r[i + 8] = hi[i]; }
  return r;
}

// A-fragment (16x32 bf16): lanes 0-15 hold K=0..15; lanes 16-31 K=16..31
static __device__ inline v16bf load_a_frag(const bf16* __restrict__ base, int lh) {
  const bf16* p = base + lh * 16;
  return combine16(*(const v8bf*)(p), *(const v8bf*)(p + 8));
}

// B-fragment: lane holds 16 contiguous K at lh*16 (two 16B reads, LDS-alignment safe)
static __device__ inline v16bf load_b_frag16(const bf16* base, int lh) {
  const bf16* p = base + lh * 16;
  return combine16(*(const v8bf*)(p), *(const v8bf*)(p + 8));
}

#define WMMA_BF16(A, B, C) \
  __builtin_amdgcn_wmma_f32_16x16x32_bf16(false, (A), false, (B), (short)0, (C), false, false)

// 16-byte global -> LDS copy (async on CDNA5; staged fallback otherwise)
static __device__ inline void copy16_g2l(const bf16* g, bf16* l) {
#ifdef HAVE_ASYNC
  typedef int v4i __attribute__((vector_size(16)));
  typedef __attribute__((address_space(1))) v4i* gptr_t;
  typedef __attribute__((address_space(3))) v4i* lptr_t;
  __builtin_amdgcn_global_load_async_to_lds_b128((gptr_t)g, (lptr_t)l, 0, 0);
#else
  *(v8bf*)l = *(const v8bf*)g;
#endif
}

static __device__ inline void wait_async_all() {
#ifdef HAVE_ASYNC
#if __has_builtin(__builtin_amdgcn_s_wait_asynccnt)
  __builtin_amdgcn_s_wait_asynccnt(0);
#else
  asm volatile("s_wait_asynccnt 0" ::: "memory");
#endif
#endif
}

// ---------------------------------------------------------------------------
// Conversion / layout kernels
// ---------------------------------------------------------------------------

__global__ __launch_bounds__(256) void k_cvt_bf16(const float* __restrict__ src,
                                                  bf16* __restrict__ dst, int n) {
  int idx = blockIdx.x * 256 + threadIdx.x;
  int stride = gridDim.x * 256;
  for (; idx < n; idx += stride) dst[idx] = (bf16)src[idx];
}

// src: K x N fp32 row-major -> dst: N x K bf16 row-major (K,N multiples of 32)
__global__ __launch_bounds__(256) void k_transpose_bf16(const float* __restrict__ src,
                                                        bf16* __restrict__ dst,
                                                        int K, int N) {
  __shared__ float tile[32][33];
  const int k0 = blockIdx.x * 32, n0 = blockIdx.y * 32;
  const int tx = threadIdx.x, ty = threadIdx.y;  // 32 x 8
#pragma unroll
  for (int r = ty; r < 32; r += 8)
    tile[r][tx] = src[(size_t)(k0 + r) * N + (n0 + tx)];
  __syncthreads();
#pragma unroll
  for (int r = ty; r < 32; r += 8)
    dst[(size_t)(n0 + r) * K + (k0 + tx)] = (bf16)tile[tx][r];
}

// src: K x 16 fp32 -> dst: 16 x K bf16
__global__ __launch_bounds__(256) void k_transpose_loraA(const float* __restrict__ src,
                                                         bf16* __restrict__ dst, int K) {
  int idx = blockIdx.x * 256 + threadIdx.x;
  if (idx >= 16 * K) return;
  int n = idx / K, k = idx - n * K;
  dst[idx] = (bf16)src[(size_t)k * 16 + n];
}

// src: 16 x N fp32 -> dst: N x 32 bf16, columns 16..31 zero-padded
__global__ __launch_bounds__(256) void k_pad_loraB(const float* __restrict__ src,
                                                   bf16* __restrict__ dst, int N) {
  int idx = blockIdx.x * 256 + threadIdx.x;
  if (idx >= N * 32) return;
  int n = idx >> 5, r = idx & 31;
  dst[idx] = (r < 16) ? (bf16)src[(size_t)r * N + n] : (bf16)0.0f;
}

// ---------------------------------------------------------------------------
// t = Xb[M,K] @ loraA^T -> t_pad[M,32] bf16 (K-padded with zeros)
// ---------------------------------------------------------------------------
__global__ __launch_bounds__(256) void k_lora_a_gemm(const bf16* __restrict__ Xb,
                                                     const bf16* __restrict__ laT,
                                                     bf16* __restrict__ t_pad, int K) {
  const int wave = threadIdx.x >> 5;
  const int lane = threadIdx.x & 31;
  const int lr = lane & 15, lh = lane >> 4;
  const int rowBase = blockIdx.x * 128 + wave * 16;

  v8f acc = {};
  for (int kb = 0; kb < K; kb += 32) {
    v16bf a = load_a_frag(Xb + (size_t)(rowBase + lr) * K + kb, lh);
    v16bf b = *(const v16bf*)(laT + (size_t)lr * K + kb + lh * 16);
    acc = WMMA_BF16(a, b, acc);
  }
#pragma unroll
  for (int v = 0; v < 8; ++v) {
    int row = rowBase + v + lh * 8;
    t_pad[(size_t)row * 32 + lr]      = (bf16)acc[v];
    t_pad[(size_t)row * 32 + 16 + lr] = (bf16)0.0f;
  }
}

// ---------------------------------------------------------------------------
// Fused gate+up: x3 = relu(x@Wg + bg + LoRA_g) * (x@Wu + bu + LoRA_u)
// block tile 128x128, 8 waves (2M x 4N), wave tile 64x32 per matrix
// double-buffered async-to-LDS staging
// ---------------------------------------------------------------------------
#define LSTR 40  // LDS row stride in elements (32 + 8 pad -> 80B rows)

__global__ __launch_bounds__(256) void k_fused_gate_up(
    const bf16* __restrict__ x1b, const bf16* __restrict__ wgT,
    const bf16* __restrict__ wuT, const float* __restrict__ b_gate,
    const float* __restrict__ b_up, const bf16* __restrict__ tg_pad,
    const bf16* __restrict__ lgb_pad, const bf16* __restrict__ tu_pad,
    const bf16* __restrict__ lub_pad, bf16* __restrict__ x3b) {
  __shared__ bf16 sA[2][128 * LSTR];
  __shared__ bf16 sBg[2][128 * LSTR];
  __shared__ bf16 sBu[2][128 * LSTR];

  const int tid = threadIdx.x;
  const int wave = tid >> 5, lane = tid & 31;
  const int lr = lane & 15, lh = lane >> 4;
  const int waveM = wave >> 2;  // 0..1 -> 64 rows
  const int waveN = wave & 3;   // 0..3 -> 32 cols
  const int blkRow = blockIdx.x * 128;
  const int blkCol = blockIdx.y * 128;

  v8f accG[4][2] = {};
  v8f accU[4][2] = {};

  // 128 rows x 4 16B-chunks = 512 chunks per tile; 2 per thread per tile
  auto copy_tiles = [&](int p, int kb) {
#pragma unroll
    for (int i = 0; i < 2; ++i) {
      const int c = tid + i * 256;
      const int row = c >> 2;
      const int ch = (c & 3) * 8;  // element offset within the 32-wide K slab
      copy16_g2l(x1b + (size_t)(blkRow + row) * DIM + kb + ch,
                 &sA[p][row * LSTR + ch]);
      copy16_g2l(wgT + (size_t)(blkCol + row) * DIM + kb + ch,
                 &sBg[p][row * LSTR + ch]);
      copy16_g2l(wuT + (size_t)(blkCol + row) * DIM + kb + ch,
                 &sBu[p][row * LSTR + ch]);
    }
  };

  // B fragments held (shared across M-subtiles); A fragment loaded per i to
  // keep peak register liveness low (acc = 128 VGPRs already).
  auto compute = [&](int p) {
    v16bf bg[2], bu[2];
#pragma unroll
    for (int j = 0; j < 2; ++j) {
      bg[j] = load_b_frag16(&sBg[p][(waveN * 32 + j * 16 + lr) * LSTR], lh);
      bu[j] = load_b_frag16(&sBu[p][(waveN * 32 + j * 16 + lr) * LSTR], lh);
    }
#pragma unroll
    for (int i = 0; i < 4; ++i) {
      v16bf a = load_a_frag(&sA[p][(waveM * 64 + i * 16 + lr) * LSTR], lh);
#pragma unroll
      for (int j = 0; j < 2; ++j) {
        accG[i][j] = WMMA_BF16(a, bg[j], accG[i][j]);
        accU[i][j] = WMMA_BF16(a, bu[j], accU[i][j]);
      }
    }
  };

  copy_tiles(0, 0);
  wait_async_all();
  __syncthreads();

  const int NT = DIM / 32;
  for (int kt = 0; kt < NT; ++kt) {
    const int p = kt & 1;
    if (kt + 1 < NT) copy_tiles(p ^ 1, (kt + 1) * 32);
    compute(p);
    wait_async_all();
    __syncthreads();
  }

  // rank-16 LoRA as one K=32 (zero-padded) WMMA per tile per matrix
#pragma unroll
  for (int i = 0; i < 4; ++i) {
    const int rowA = blkRow + waveM * 64 + i * 16 + lr;
    v16bf ag = load_a_frag(tg_pad + (size_t)rowA * 32, lh);
    v16bf au = load_a_frag(tu_pad + (size_t)rowA * 32, lh);
#pragma unroll
    for (int j = 0; j < 2; ++j) {
      size_t co = (size_t)(blkCol + waveN * 32 + j * 16 + lr) * 32 + lh * 16;
      accG[i][j] = WMMA_BF16(ag, *(const v16bf*)(lgb_pad + co), accG[i][j]);
      accU[i][j] = WMMA_BF16(au, *(const v16bf*)(lub_pad + co), accU[i][j]);
    }
  }

  // bias, relu-gate, store x3 as bf16
#pragma unroll
  for (int j = 0; j < 2; ++j) {
    const int col = blkCol + waveN * 32 + j * 16 + lr;
    const float bgv = b_gate[col];
    const float buv = b_up[col];
#pragma unroll
    for (int i = 0; i < 4; ++i) {
      const int row0 = blkRow + waveM * 64 + i * 16 + lh * 8;
#pragma unroll
      for (int v = 0; v < 8; ++v) {
        float y1 = accG[i][j][v] + bgv;
        float y2 = accU[i][j][v] + buv;
        float x3 = (y1 > 0.0f ? y1 : 0.0f) * y2;
        x3b[(size_t)(row0 + v) * FF + col] = (bf16)x3;
      }
    }
  }
}

// ---------------------------------------------------------------------------
// Down projection: y3 = x3 @ Wd + bd + LoRA_d (fp32 out)
// block tile 128x128, wave tile 64x32, double-buffered async-to-LDS staging
// ---------------------------------------------------------------------------
__global__ __launch_bounds__(256) void k_down_proj(
    const bf16* __restrict__ x3b, const bf16* __restrict__ wdT,
    const float* __restrict__ b_down, const bf16* __restrict__ td_pad,
    const bf16* __restrict__ ldb_pad, float* __restrict__ out) {
  __shared__ bf16 sA[2][128 * LSTR];
  __shared__ bf16 sB[2][128 * LSTR];

  const int tid = threadIdx.x;
  const int wave = tid >> 5, lane = tid & 31;
  const int lr = lane & 15, lh = lane >> 4;
  const int waveM = wave >> 2;
  const int waveN = wave & 3;
  const int blkRow = blockIdx.x * 128;
  const int blkCol = blockIdx.y * 128;

  v8f acc[4][2] = {};

  auto copy_tiles = [&](int p, int kb) {
#pragma unroll
    for (int i = 0; i < 2; ++i) {
      const int c = tid + i * 256;
      const int row = c >> 2;
      const int ch = (c & 3) * 8;
      copy16_g2l(x3b + (size_t)(blkRow + row) * FF + kb + ch,
                 &sA[p][row * LSTR + ch]);
      copy16_g2l(wdT + (size_t)(blkCol + row) * FF + kb + ch,
                 &sB[p][row * LSTR + ch]);
    }
  };

  auto compute = [&](int p) {
    v16bf b[2];
#pragma unroll
    for (int j = 0; j < 2; ++j)
      b[j] = load_b_frag16(&sB[p][(waveN * 32 + j * 16 + lr) * LSTR], lh);
#pragma unroll
    for (int i = 0; i < 4; ++i) {
      v16bf a = load_a_frag(&sA[p][(waveM * 64 + i * 16 + lr) * LSTR], lh);
#pragma unroll
      for (int j = 0; j < 2; ++j) acc[i][j] = WMMA_BF16(a, b[j], acc[i][j]);
    }
  };

  copy_tiles(0, 0);
  wait_async_all();
  __syncthreads();

  const int NT = FF / 32;
  for (int kt = 0; kt < NT; ++kt) {
    const int p = kt & 1;
    if (kt + 1 < NT) copy_tiles(p ^ 1, (kt + 1) * 32);
    compute(p);
    wait_async_all();
    __syncthreads();
  }

#pragma unroll
  for (int i = 0; i < 4; ++i) {
    const int rowA = blkRow + waveM * 64 + i * 16 + lr;
    v16bf ad = load_a_frag(td_pad + (size_t)rowA * 32, lh);
#pragma unroll
    for (int j = 0; j < 2; ++j) {
      size_t co = (size_t)(blkCol + waveN * 32 + j * 16 + lr) * 32 + lh * 16;
      acc[i][j] = WMMA_BF16(ad, *(const v16bf*)(ldb_pad + co), acc[i][j]);
    }
  }

#pragma unroll
  for (int j = 0; j < 2; ++j) {
    const int col = blkCol + waveN * 32 + j * 16 + lr;
    const float bdv = b_down[col];
#pragma unroll
    for (int i = 0; i < 4; ++i) {
      const int row0 = blkRow + waveM * 64 + i * 16 + lh * 8;
#pragma unroll
      for (int v = 0; v < 8; ++v)
        out[(size_t)(row0 + v) * DIM + col] = acc[i][j][v] + bdv;
    }
  }
}

// ---------------------------------------------------------------------------
// host-side launch
// ---------------------------------------------------------------------------
extern "C" void kernel_launch(void* const* d_in, const int* in_sizes, int n_in,
                              void* d_out, int out_size, void* d_ws, size_t ws_size,
                              hipStream_t stream) {
  const float* x1     = (const float*)d_in[0];
  const float* w_gate = (const float*)d_in[1];
  const float* b_gate = (const float*)d_in[2];
  const float* w_gla  = (const float*)d_in[3];
  const float* w_glb  = (const float*)d_in[4];
  const float* w_up   = (const float*)d_in[5];
  const float* b_up   = (const float*)d_in[6];
  const float* w_ula  = (const float*)d_in[7];
  const float* w_ulb  = (const float*)d_in[8];
  const float* w_down = (const float*)d_in[9];
  const float* b_down = (const float*)d_in[10];
  const float* w_dla  = (const float*)d_in[11];
  const float* w_dlb  = (const float*)d_in[12];
  float* out = (float*)d_out;

  char* ws = (char*)d_ws;
  size_t off = 0;
  auto take = [&](size_t bytes) -> char* {
    char* p = ws + off;
    off += (bytes + 255) & ~(size_t)255;
    return p;
  };

  bf16* x1b     = (bf16*)take((size_t)M_TOK * DIM * 2);   //  32 MB
  bf16* wgT     = (bf16*)take((size_t)FF * DIM * 2);      //  32 MB
  bf16* wuT     = (bf16*)take((size_t)FF * DIM * 2);      //  32 MB
  bf16* wdT     = (bf16*)take((size_t)DIM * FF * 2);      //  32 MB
  bf16* x3b     = (bf16*)take((size_t)M_TOK * FF * 2);    // 128 MB
  bf16* lgaT    = (bf16*)take((size_t)RLORA * DIM * 2);
  bf16* luaT    = (bf16*)take((size_t)RLORA * DIM * 2);
  bf16* ldaT    = (bf16*)take((size_t)RLORA * FF * 2);
  bf16* lgb_pad = (bf16*)take((size_t)FF * 32 * 2);
  bf16* lub_pad = (bf16*)take((size_t)FF * 32 * 2);
  bf16* ldb_pad = (bf16*)take((size_t)DIM * 32 * 2);
  bf16* tg_pad  = (bf16*)take((size_t)M_TOK * 32 * 2);
  bf16* tu_pad  = (bf16*)take((size_t)M_TOK * 32 * 2);
  bf16* td_pad  = (bf16*)take((size_t)M_TOK * 32 * 2);
  (void)ws_size; (void)in_sizes; (void)n_in; (void)out_size;

  const dim3 blk256(256);
  const dim3 blkT(32, 8);

  // --- layout / precision conversion ---
  k_cvt_bf16<<<dim3(32768), blk256, 0, stream>>>(x1, x1b, M_TOK * DIM);
  k_transpose_bf16<<<dim3(DIM / 32, FF / 32), blkT, 0, stream>>>(w_gate, wgT, DIM, FF);
  k_transpose_bf16<<<dim3(DIM / 32, FF / 32), blkT, 0, stream>>>(w_up, wuT, DIM, FF);
  k_transpose_bf16<<<dim3(FF / 32, DIM / 32), blkT, 0, stream>>>(w_down, wdT, FF, DIM);
  k_transpose_loraA<<<dim3((16 * DIM + 255) / 256), blk256, 0, stream>>>(w_gla, lgaT, DIM);
  k_transpose_loraA<<<dim3((16 * DIM + 255) / 256), blk256, 0, stream>>>(w_ula, luaT, DIM);
  k_transpose_loraA<<<dim3((16 * FF + 255) / 256), blk256, 0, stream>>>(w_dla, ldaT, FF);
  k_pad_loraB<<<dim3((FF * 32 + 255) / 256), blk256, 0, stream>>>(w_glb, lgb_pad, FF);
  k_pad_loraB<<<dim3((FF * 32 + 255) / 256), blk256, 0, stream>>>(w_ulb, lub_pad, FF);
  k_pad_loraB<<<dim3((DIM * 32 + 255) / 256), blk256, 0, stream>>>(w_dlb, ldb_pad, DIM);

  // --- LoRA-A projections (t = x @ A, K-padded bf16) ---
  k_lora_a_gemm<<<dim3(M_TOK / 128), blk256, 0, stream>>>(x1b, lgaT, tg_pad, DIM);
  k_lora_a_gemm<<<dim3(M_TOK / 128), blk256, 0, stream>>>(x1b, luaT, tu_pad, DIM);

  // --- fused gate+up -> x3 (bf16) ---
  k_fused_gate_up<<<dim3(M_TOK / 128, FF / 128), blk256, 0, stream>>>(
      x1b, wgT, wuT, b_gate, b_up, tg_pad, lgb_pad, tu_pad, lub_pad, x3b);

  // --- LoRA-A for down, then down projection -> out (fp32) ---
  k_lora_a_gemm<<<dim3(M_TOK / 128), blk256, 0, stream>>>(x3b, ldaT, td_pad, FF);
  k_down_proj<<<dim3(M_TOK / 128, DIM / 128), blk256, 0, stream>>>(
      x3b, wdT, b_down, td_pad, ldb_pad, out);
}
